// Temporal_CL_Regularizer_48232482734699
// MI455X (gfx1250) — matrix operand order repaired and checked by reading, not compiled, not run on hardware
//
#include <hip/hip_runtime.h>
#include <hip/hip_bf16.h>

typedef _Float16 half4_t __attribute__((ext_vector_type(4)));
typedef _Float16 half8_t __attribute__((ext_vector_type(8)));
typedef _Float16 v16h    __attribute__((ext_vector_type(16)));
typedef float    v8f     __attribute__((ext_vector_type(8)));

#define NROWS 8192
#define DDIM  128
#define WEIGHTC 0.01f

union FragU { v16h v; half8_t h[2]; };

// ---------------------------------------------------------------------------
// Kernel 1: row L2-normalize, emit f16 matrix. One wave (32 lanes) per row.
// ---------------------------------------------------------------------------
__global__ __launch_bounds__(256) void tclr_normalize(const float* __restrict__ in,
                                                      _Float16* __restrict__ fh) {
    const int tid  = threadIdx.x;
    const int lane = tid & 31;
    const int wave = tid >> 5;
    const int row  = blockIdx.x * 8 + wave;

    const float4 x = *(const float4*)(in + (size_t)row * DDIM + lane * 4);
    float ss = x.x * x.x + x.y * x.y + x.z * x.z + x.w * x.w;
#pragma unroll
    for (int m = 16; m >= 1; m >>= 1) ss += __shfl_xor(ss, m, 32);
    const float inv = 1.0f / fmaxf(sqrtf(ss), 1e-12f);

    half4_t o;
    o.x = (_Float16)(x.x * inv);
    o.y = (_Float16)(x.y * inv);
    o.z = (_Float16)(x.z * inv);
    o.w = (_Float16)(x.w * inv);
    *(half4_t*)(fh + (size_t)row * DDIM + lane * 4) = o;
}

// ---------------------------------------------------------------------------
// Kernel 2: Gram tiles via v_wmma_f32_16x16x32_f16, fused exp + banded sums.
//   U[i] = sum_{j>=i+2} exp(sim[i,j]),  L[i] = sum_{j<=i-2} exp(sim[i,j])
// WG = 64 rows (4 tile-rows) x 8 waves (4 tile-rows x 2 column halves).
// Software-pipelined global->reg->LDS staging of the 128-column B block;
// k-outer / tile-inner WMMA order gives 4 independent accumulator chains.
// CU mode: 256-thread WG = 8 waves on 2 SIMD32s -> 4 waves/SIMD floor;
// amdgpu_waves_per_eu(4,4) raises the VGPR budget to file/4 (~256) so the
// staging registers and all four B fragments stay live (no scratch spills).
// ---------------------------------------------------------------------------
#define LDSB_STRIDE 136   // f16 units; 272 B rows -> conflict-free frag reads

__global__ __launch_bounds__(256)
__attribute__((amdgpu_waves_per_eu(4, 4)))
void tclr_gram(const _Float16* __restrict__ fh,
               float* __restrict__ U,
               float* __restrict__ L) {
    __shared__ _Float16 Bs[128 * LDSB_STRIDE];   // ~34 KB
    __shared__ float sU[64];
    __shared__ float sL[64];

    const int tid  = threadIdx.x;
    const int lane = tid & 31;
    const int wave = tid >> 5;
    const int hf   = lane >> 4;      // wave half (0/1)
    const int l15  = lane & 15;
    const int tr   = wave >> 1;      // tile-row within block (0..3)
    const int ch   = wave & 1;       // column half (0/1)
    const int rowbase = blockIdx.x * 64;
    const int rowLo   = rowbase + tr * 16;

    if (tid < 64) { sU[tid] = 0.0f; sL[tid] = 0.0f; }

    // Persistent A fragments: this wave's 16 rows x full K=128 (4 chunks of 32).
    // A layout (16-bit 16x32): elems 0..7 -> K = 8*hf + e ; elems 8..15 -> 16 + 8*hf + (e-8)
    FragU a[4];
    {
        const _Float16* arow = fh + (size_t)(rowLo + l15) * DDIM;
#pragma unroll
        for (int c = 0; c < 4; ++c) {
            a[c].h[0] = *(const half8_t*)(arow + 32 * c + 8 * hf);
            a[c].h[1] = *(const half8_t*)(arow + 32 * c + 16 + 8 * hf);
        }
    }

    float sumU[8], sumL[8];
#pragma unroll
    for (int r = 0; r < 8; ++r) { sumU[r] = 0.0f; sumL[r] = 0.0f; }

    // staging geometry: 2 threads per B row, 128 contiguous bytes each
    const int r_st  = tid >> 1;
    const int hc_st = (tid & 1) * 64;
    uint4* const dstLds = (uint4*)((char*)Bs + r_st * (LDSB_STRIDE * 2) + hc_st * 2);

    uint4 stage[8];
    {   // prologue: stage block 0
        const uint4* src = (const uint4*)(fh + (size_t)r_st * DDIM + hc_st);
#pragma unroll
        for (int q = 0; q < 8; ++q) stage[q] = src[q];
#pragma unroll
        for (int q = 0; q < 8; ++q) dstLds[q] = stage[q];
    }
    __syncthreads();

    for (int b = 0; b < NROWS / 128; ++b) {
        const int jb = b * 128;

        if (b + 1 < NROWS / 128) {   // issue next block's global loads early
            const uint4* src = (const uint4*)(fh + (size_t)(jb + 128 + r_st) * DDIM + hc_st);
#pragma unroll
            for (int q = 0; q < 8; ++q) stage[q] = src[q];
        }
        if (b + 2 < NROWS / 128) {
            __builtin_prefetch((const char*)(fh + (size_t)(jb + 256 + r_st) * DDIM + hc_st), 0, 1);
        }

        // ---- compute: 4 independent accumulator chains, batched LDS feeds ----
        v8f acc[4];
#pragma unroll
        for (int t = 0; t < 4; ++t) acc[t] = (v8f){0.f,0.f,0.f,0.f,0.f,0.f,0.f,0.f};

#pragma unroll
        for (int c = 0; c < 4; ++c) {
            FragU bt[4];
#pragma unroll
            for (int t = 0; t < 4; ++t) {
                // B layout (16-bit 32x16): elems 0..7 -> K = 8*hf + e ; elems 8..15 -> 16 + 8*hf + (e-8)
                const _Float16* brow = Bs + (size_t)((ch * 4 + t) * 16 + l15) * LDSB_STRIDE;
                bt[t].h[0] = *(const half8_t*)(brow + 32 * c + 8 * hf);
                bt[t].h[1] = *(const half8_t*)(brow + 32 * c + 16 + 8 * hf);
            }
#pragma unroll
            for (int t = 0; t < 4; ++t) {
                acc[t] = __builtin_amdgcn_wmma_f32_16x16x32_f16(
                            false, a[c].v, false, bt[t].v, (short)0, acc[t], false, false);
            }
        }

        // ---- epilogue: exp + banded accumulation; band masks only near diag ----
#pragma unroll
        for (int t = 0; t < 4; ++t) {
            const int colLo = jb + (ch * 4 + t) * 16;   // wave-uniform
            const int ncol  = colLo + l15;
            if (colLo >= rowLo + 17) {                  // tile entirely above band
#pragma unroll
                for (int r = 0; r < 8; ++r) sumU[r] += __expf(acc[t][r]);
            } else if (colLo <= rowLo - 17) {           // tile entirely below band
#pragma unroll
                for (int r = 0; r < 8; ++r) sumL[r] += __expf(acc[t][r]);
            } else {                                    // diagonal-straddling tile
#pragma unroll
                for (int r = 0; r < 8; ++r) {
                    const int i = rowLo + 8 * hf + r;
                    const float e = __expf(acc[t][r]);
                    if (ncol >= i + 2) sumU[r] += e;
                    if (ncol <= i - 2) sumL[r] += e;
                }
            }
        }

        __syncthreads();                    // everyone done reading Bs
        if (b + 1 < NROWS / 128) {
#pragma unroll
            for (int q = 0; q < 8; ++q) dstLds[q] = stage[q];
        }
        __syncthreads();                    // Bs holds block b+1
    }

    // reduce over the 16 lanes of each half (they cover distinct columns)
#pragma unroll
    for (int m = 1; m <= 8; m <<= 1) {
#pragma unroll
        for (int r = 0; r < 8; ++r) {
            sumU[r] += __shfl_xor(sumU[r], m, 32);
            sumL[r] += __shfl_xor(sumL[r], m, 32);
        }
    }
    if (l15 == 0) {   // lanes 0 and 16: combine the two column-half waves in LDS
        const int base = tr * 16 + 8 * hf;
#pragma unroll
        for (int r = 0; r < 8; ++r) {
            atomicAdd(&sU[base + r], sumU[r]);
            atomicAdd(&sL[base + r], sumL[r]);
        }
    }
    __syncthreads();
    if (tid < 64) {
        U[rowbase + tid] = sU[tid];
        L[rowbase + tid] = sL[tid];
    }
}

// ---------------------------------------------------------------------------
// Kernel 3: superdiagonal p_k = <f[k], f[k+1]> in fp32 (from the f16 data).
// ---------------------------------------------------------------------------
__global__ __launch_bounds__(256) void tclr_pos(const _Float16* __restrict__ fh,
                                                float* __restrict__ P) {
    const int tid  = threadIdx.x;
    const int lane = tid & 31;
    const int wave = tid >> 5;
    const int k = blockIdx.x * 8 + wave;
    if (k >= NROWS - 1) return;

    const half4_t av = *(const half4_t*)(fh + (size_t)k * DDIM + lane * 4);
    const half4_t bv = *(const half4_t*)(fh + (size_t)(k + 1) * DDIM + lane * 4);
    float d = (float)av.x * (float)bv.x + (float)av.y * (float)bv.y +
              (float)av.z * (float)bv.z + (float)av.w * (float)bv.w;
#pragma unroll
    for (int m = 16; m >= 1; m >>= 1) d += __shfl_xor(d, m, 32);
    if (lane == 0) P[k] = d;
}

// ---------------------------------------------------------------------------
// Kernel 4: loss = -W/N * sum_k [ log(exp(p_k) + U[k] + L[k+1]) - p_k ]
// ---------------------------------------------------------------------------
__global__ __launch_bounds__(256) void tclr_loss(const float* __restrict__ U,
                                                 const float* __restrict__ L,
                                                 const float* __restrict__ P,
                                                 float* __restrict__ out) {
    __shared__ float red[256];
    float s = 0.0f;
    for (int k = threadIdx.x; k < NROWS - 1; k += 256) {
        const float p = P[k];
        const float S = U[k] + L[k + 1];
        s += logf(__expf(p) + S) - p;
    }
    red[threadIdx.x] = s;
    __syncthreads();
#pragma unroll
    for (int off = 128; off >= 1; off >>= 1) {
        if (threadIdx.x < off) red[threadIdx.x] += red[threadIdx.x + off];
        __syncthreads();
    }
    if (threadIdx.x == 0) out[0] = -WEIGHTC * (red[0] / (float)NROWS);
}

// ---------------------------------------------------------------------------
extern "C" void kernel_launch(void* const* d_in, const int* in_sizes, int n_in,
                              void* d_out, int out_size, void* d_ws, size_t ws_size,
                              hipStream_t stream) {
    const float* factor = (const float*)d_in[0];
    float* out = (float*)d_out;

    char* ws = (char*)d_ws;
    _Float16* fh = (_Float16*)ws;                               // 2 MB
    float* U = (float*)(ws + (size_t)NROWS * DDIM * sizeof(_Float16));
    float* L = U + NROWS;
    float* P = L + NROWS;

    tclr_normalize<<<NROWS / 8, 256, 0, stream>>>(factor, fh);
    tclr_gram<<<NROWS / 64, 256, 0, stream>>>(fh, U, L);
    tclr_pos<<<NROWS / 8, 256, 0, stream>>>(fh, P);
    tclr_loss<<<1, 256, 0, stream>>>(U, L, P, out);
}
